// LanguageModel_45088566674140
// MI455X (gfx1250) — compile-verified, matrix operand order checked by reading, not run on hardware
//
#include <hip/hip_runtime.h>

typedef __attribute__((ext_vector_type(16))) __bf16 bf16x16;
typedef __attribute__((ext_vector_type(8)))  __bf16 bf16x8;
typedef __attribute__((ext_vector_type(8)))  float  f32x8;
typedef __attribute__((ext_vector_type(2)))  float  f32x2;

#define B_   64
#define S_   256
#define V_   50257
#define E_   512
#define H_   1024
#define G4H_ 4096

// ---------------------------------------------------------------------------
// Prep: embedding lookup + cast to bf16.  xe[b, s, e] = bf16(embed[x[b,s], e])
// ---------------------------------------------------------------------------
__global__ __launch_bounds__(256)
void embed_cast_kernel(const int* __restrict__ x, const float* __restrict__ embed,
                       __bf16* __restrict__ xe) {
  long i = (long)blockIdx.x * blockDim.x + threadIdx.x;   // < B*S*E
  int  e  = (int)(i % E_);
  long bs = i / E_;
  int  tok = x[bs];
  xe[i] = (__bf16)embed[(long)tok * E_ + e];
}

// ---------------------------------------------------------------------------
// Prep: Wt[n, k] = bf16( k < K0 ? Wa[k, n] : Wb[k-K0, n] ), n in [0,4096)
// Tiled 16x16 transpose through LDS so both sides stay coalesced.
// ---------------------------------------------------------------------------
__global__ __launch_bounds__(256)
void transpose_cat_bf16(const float* __restrict__ Wa, const float* __restrict__ Wb,
                        int K0, int K, __bf16* __restrict__ Wt) {
  __shared__ float tile[16][17];
  int k0 = blockIdx.x * 16;
  int n0 = blockIdx.y * 16;
  int tx = threadIdx.x, ty = threadIdx.y;
  int k = k0 + ty, n = n0 + tx;
  float v = (k < K0) ? Wa[(long)k * G4H_ + n] : Wb[(long)(k - K0) * G4H_ + n];
  tile[ty][tx] = v;
  __syncthreads();
  Wt[(long)(n0 + ty) * K + (k0 + tx)] = (__bf16)tile[tx][ty];
}

// ---------------------------------------------------------------------------
// Prep: zero the recurrent state region (c0, c1, h0bf[2], h1bf[2])
// ---------------------------------------------------------------------------
__global__ __launch_bounds__(256)
void zero_kernel(float* __restrict__ p, long n) {
  long i = (long)blockIdx.x * blockDim.x + threadIdx.x;
  if (i < n) p[i] = 0.0f;
}

// ---------------------------------------------------------------------------
// One LSTM layer for one timestep, fused GEMM + cell.
//   z = [A0 | A1] @ Wt^T + bias     (M=64, N=4096, K = K0 + K1, compile-time)
// Grid: (32 n-chunks of 32 cols, 4 m-tiles), block = 128 = 4 waves (1/gate).
// Each wave computes TWO 16x16 tiles of its gate (2-way N blocking) so the
// A tile is reused for two WMMAs.  Gates meet in LDS for the cell update.
// ---------------------------------------------------------------------------
template <int K0, int K1>
__global__ __launch_bounds__(128)
void lstm_step_kernel(const __bf16* __restrict__ A0, int lda0,
                      const __bf16* __restrict__ A1, int lda1,
                      const __bf16* __restrict__ Wt,    // [4096, K0+K1] bf16
                      const float*  __restrict__ bias,  // [4096]
                      float*        __restrict__ c,     // [64, 1024] in/out
                      float*        __restrict__ hf,    // [64, 1024] f32 out
                      __bf16*       __restrict__ hb) {  // [64, 1024] bf16 out
  constexpr int K = K0 + K1;
  __shared__ float zs[4][16][33];
  const int lane   = threadIdx.x & 31;
  const int w      = threadIdx.x >> 5;        // gate: 0=i 1=f 2=g 3=o
  const int m0     = blockIdx.y * 16;         // batch-row tile
  const int nchunk = blockIdx.x;              // 0..31, 32 hidden cols each
  const int khalf  = (lane >> 4) * 8;         // ISA 16-bit A/B lane split
  const int row    = m0 + (lane & 15);
  const int ncol0  = w * H_ + nchunk * 32 + (lane & 15);
  const int ncol1  = ncol0 + 16;

  // Per-lane base pointers (khalf folded in) so loops use immediate offsets.
  const __bf16* a0p = A0 + (long)row * lda0 + khalf;
  const __bf16* a1p = A1 + (long)row * lda1 + khalf;
  const __bf16* w0p = Wt + (long)ncol0 * K + khalf;
  const __bf16* w1p = Wt + (long)ncol1 * K + khalf;

  f32x8 acc0 = {}, acc1 = {};

#pragma unroll 8
  for (int kb = 0; kb < K0; kb += 32) {
    bf16x8 alo = *(const bf16x8*)(a0p + kb);
    bf16x8 ahi = *(const bf16x8*)(a0p + kb + 16);
    bf16x8 b0l = *(const bf16x8*)(w0p + kb);
    bf16x8 b0h = *(const bf16x8*)(w0p + kb + 16);
    bf16x8 b1l = *(const bf16x8*)(w1p + kb);
    bf16x8 b1h = *(const bf16x8*)(w1p + kb + 16);
    bf16x16 av  = __builtin_shufflevector(alo, ahi, 0,1,2,3,4,5,6,7,8,9,10,11,12,13,14,15);
    bf16x16 bv0 = __builtin_shufflevector(b0l, b0h, 0,1,2,3,4,5,6,7,8,9,10,11,12,13,14,15);
    bf16x16 bv1 = __builtin_shufflevector(b1l, b1h, 0,1,2,3,4,5,6,7,8,9,10,11,12,13,14,15);
    acc0 = __builtin_amdgcn_wmma_f32_16x16x32_bf16(false, av, false, bv0,
                                                   (short)0, acc0, false, false);
    acc1 = __builtin_amdgcn_wmma_f32_16x16x32_bf16(false, av, false, bv1,
                                                   (short)0, acc1, false, false);
  }
#pragma unroll 8
  for (int kb = 0; kb < K1; kb += 32) {
    bf16x8 alo = *(const bf16x8*)(a1p + kb);
    bf16x8 ahi = *(const bf16x8*)(a1p + kb + 16);
    bf16x8 b0l = *(const bf16x8*)(w0p + K0 + kb);
    bf16x8 b0h = *(const bf16x8*)(w0p + K0 + kb + 16);
    bf16x8 b1l = *(const bf16x8*)(w1p + K0 + kb);
    bf16x8 b1h = *(const bf16x8*)(w1p + K0 + kb + 16);
    bf16x16 av  = __builtin_shufflevector(alo, ahi, 0,1,2,3,4,5,6,7,8,9,10,11,12,13,14,15);
    bf16x16 bv0 = __builtin_shufflevector(b0l, b0h, 0,1,2,3,4,5,6,7,8,9,10,11,12,13,14,15);
    bf16x16 bv1 = __builtin_shufflevector(b1l, b1h, 0,1,2,3,4,5,6,7,8,9,10,11,12,13,14,15);
    acc0 = __builtin_amdgcn_wmma_f32_16x16x32_bf16(false, av, false, bv0,
                                                   (short)0, acc0, false, false);
    acc1 = __builtin_amdgcn_wmma_f32_16x16x32_bf16(false, av, false, bv1,
                                                   (short)0, acc1, false, false);
  }

  // D layout: lane holds col n = lane&15, VGPR r holds row m = r + 8*(lane>>4)
  const float bz0  = bias[ncol0];
  const float bz1  = bias[ncol1];
  const int   mrow = 8 * (lane >> 4);
#pragma unroll
  for (int r = 0; r < 8; ++r) {
    zs[w][mrow + r][(lane & 15)]      = acc0[r] + bz0;
    zs[w][mrow + r][(lane & 15) + 16] = acc1[r] + bz1;
  }
  __syncthreads();

  // LSTM cell: 512 elements per block, 128 threads -> 4 each.
  for (int e = threadIdx.x; e < 512; e += 128) {
    int m = e >> 5, n = e & 31;
    float zi = zs[0][m][n], zf = zs[1][m][n], zg = zs[2][m][n], zo = zs[3][m][n];
    long idx = (long)(m0 + m) * H_ + (nchunk * 32 + n);
    float cv = c[idx];
    float si = 1.0f / (1.0f + __expf(-zi));
    float sf = 1.0f / (1.0f + __expf(-zf));
    float so = 1.0f / (1.0f + __expf(-zo));
    float cn = sf * cv + si * tanhf(zg);
    float hn = so * tanhf(cn);
    c[idx]  = cn;
    hf[idx] = hn;
    hb[idx] = (__bf16)hn;
  }
}

// ---------------------------------------------------------------------------
// logits = h1 @ Wout + bout  with fp32 WMMA (16x16x4).  M=64, K=1024, N=V.
// Block = 128 threads = 4 waves; wave w handles m-tile w; grid.x tiles N.
// Tail tile handled by CLAMPING the column index (loads always valid,
// garbage lanes recompute column V-1) and masking only the final store —
// keeps the hot loop free of EXEC manipulation.
// ---------------------------------------------------------------------------
__global__ __launch_bounds__(128)
void out_gemm_kernel(const float* __restrict__ h,      // [64, 1024]
                     const float* __restrict__ Wout,   // [1024, V]
                     const float* __restrict__ bout,   // [V]
                     float* __restrict__ logits) {     // [64, V]
  const int lane = threadIdx.x & 31;
  const int m0   = (threadIdx.x >> 5) * 16;
  const int n    = blockIdx.x * 16 + (lane & 15);
  const int nc   = (n < V_) ? n : (V_ - 1);   // clamped load column
  const int kp   = (lane >> 4) * 2;           // f32 A/B lane split (K pairs)
  const int row  = m0 + (lane & 15);

  f32x8 acc = {};
  const float* hrow = h + (long)row * H_ + kp;
  const float* wp   = Wout + (long)kp * V_ + nc;
#pragma unroll 8
  for (int kb = 0; kb < H_; kb += 4) {
    f32x2 a = *(const f32x2*)(hrow + kb);
    f32x2 b;
    b.x = wp[(long)kb * V_];
    b.y = wp[(long)(kb + 1) * V_];
    acc = __builtin_amdgcn_wmma_f32_16x16x4_f32(false, a, false, b,
                                                (short)0, acc, false, false);
  }
  if (n < V_) {
    const float bz   = bout[n];
    const int   mrow = 8 * (lane >> 4);
#pragma unroll
    for (int r = 0; r < 8; ++r)
      logits[(long)(m0 + mrow + r) * V_ + n] = acc[r] + bz;
  }
}

// ---------------------------------------------------------------------------
// Row softmax: one block per batch row.
// ---------------------------------------------------------------------------
__global__ __launch_bounds__(256)
void softmax_kernel(const float* __restrict__ logits, float* __restrict__ out) {
  __shared__ float red[256];
  const int b = blockIdx.x;
  const float* lr = logits + (long)b * V_;
  float mx = -3.4e38f;
  for (int v = threadIdx.x; v < V_; v += 256) mx = fmaxf(mx, lr[v]);
  red[threadIdx.x] = mx; __syncthreads();
  for (int s = 128; s > 0; s >>= 1) {
    if (threadIdx.x < s) red[threadIdx.x] = fmaxf(red[threadIdx.x], red[threadIdx.x + s]);
    __syncthreads();
  }
  mx = red[0]; __syncthreads();
  float sum = 0.0f;
  for (int v = threadIdx.x; v < V_; v += 256) sum += __expf(lr[v] - mx);
  red[threadIdx.x] = sum; __syncthreads();
  for (int s = 128; s > 0; s >>= 1) {
    if (threadIdx.x < s) red[threadIdx.x] += red[threadIdx.x + s];
    __syncthreads();
  }
  const float inv = 1.0f / red[0];
  for (int v = threadIdx.x; v < V_; v += 256)
    out[(long)b * V_ + v] = __expf(lr[v] - mx) * inv;
}

// ---------------------------------------------------------------------------
extern "C" void kernel_launch(void* const* d_in, const int* in_sizes, int n_in,
                              void* d_out, int out_size, void* d_ws, size_t ws_size,
                              hipStream_t stream) {
  const int*   x     = (const int*)d_in[0];
  const float* embed = (const float*)d_in[1];
  const float* Wx0   = (const float*)d_in[2];
  const float* Wh0   = (const float*)d_in[3];
  const float* b0    = (const float*)d_in[4];
  const float* Wx1   = (const float*)d_in[5];
  const float* Wh1   = (const float*)d_in[6];
  const float* b1    = (const float*)d_in[7];
  const float* Wout  = (const float*)d_in[8];
  const float* bout  = (const float*)d_in[9];
  float* out = (float*)d_out;

  char* ws = (char*)d_ws;
  size_t off = 0;
  auto alloc = [&](size_t bytes) -> void* {
    void* p = ws + off;
    off = (off + bytes + 255) & ~(size_t)255;
    return p;
  };

  __bf16* xe  = (__bf16*)alloc((size_t)B_ * S_ * E_ * 2);       // 16.8 MB
  __bf16* W0t = (__bf16*)alloc((size_t)G4H_ * 1536 * 2);        // 12.6 MB
  __bf16* W1t = (__bf16*)alloc((size_t)G4H_ * 2048 * 2);        // 16.8 MB

  // recurrent state region (zeroed each call); keep contiguous
  char* state = (char*)alloc(0);
  float*  c0     = (float*)alloc((size_t)B_ * H_ * 4);
  float*  c1     = (float*)alloc((size_t)B_ * H_ * 4);
  __bf16* h0bf[2] = { (__bf16*)alloc((size_t)B_ * H_ * 2),
                      (__bf16*)alloc((size_t)B_ * H_ * 2) };
  __bf16* h1bf[2] = { (__bf16*)alloc((size_t)B_ * H_ * 2),
                      (__bf16*)alloc((size_t)B_ * H_ * 2) };
  size_t state_bytes = (size_t)((char*)alloc(0) - state);

  float* h0f    = (float*)alloc((size_t)B_ * H_ * 4);   // write-only dump
  float* h1f    = (float*)alloc((size_t)B_ * H_ * 4);
  float* logits = (float*)alloc((size_t)B_ * V_ * 4);   // 12.9 MB

  // 1) embedding lookup -> bf16
  {
    long n = (long)B_ * S_ * E_;
    embed_cast_kernel<<<(int)(n / 256), 256, 0, stream>>>(x, embed, xe);
  }
  // 2) weight transpose+cast
  transpose_cat_bf16<<<dim3(1536 / 16, G4H_ / 16), dim3(16, 16), 0, stream>>>(
      Wx0, Wh0, E_, 1536, W0t);
  transpose_cat_bf16<<<dim3(2048 / 16, G4H_ / 16), dim3(16, 16), 0, stream>>>(
      Wx1, Wh1, H_, 2048, W1t);
  // 3) zero recurrent state
  {
    long n = (long)(state_bytes / 4);
    zero_kernel<<<(int)((n + 255) / 256), 256, 0, stream>>>((float*)state, n);
  }
  // 4) sequential scan: 2 fused kernels per step
  for (int t = 0; t < S_; ++t) {
    int cur = t & 1, nxt = (t + 1) & 1;
    lstm_step_kernel<E_, H_><<<dim3(32, 4), 128, 0, stream>>>(
        xe + (long)t * E_, S_ * E_,      // A0 = x_t  (row stride S*E)
        h0bf[cur], H_,                   // A1 = h0_{t-1}
        W0t, b0, c0, h0f, h0bf[nxt]);
    lstm_step_kernel<H_, H_><<<dim3(32, 4), 128, 0, stream>>>(
        h0bf[nxt], H_,                   // A0 = h0_t
        h1bf[cur], H_,                   // A1 = h1_{t-1}
        W1t, b1, c1, h1f, h1bf[nxt]);
  }
  // 5) output head + softmax (h1f holds h1 at t = S-1)
  out_gemm_kernel<<<(V_ + 15) / 16, 128, 0, stream>>>(h1f, Wout, bout, logits);
  softmax_kernel<<<B_, 256, 0, stream>>>(logits, out);
}
